// SpatialAttention2D_22230750724172
// MI455X (gfx1250) — matrix operand order repaired and checked by reading, not compiled, not run on hardware
//
#include <hip/hip_runtime.h>

#define DEV static __device__ __forceinline__

typedef _Float16 v16h __attribute__((ext_vector_type(16)));
typedef _Float16 v8h  __attribute__((ext_vector_type(8)));
typedef float    v8f  __attribute__((ext_vector_type(8)));

union V16H { v16h v; v8h h[2]; unsigned u[8]; };

constexpr int B_  = 4;
constexpr int H_  = 48;
constexpr int W_  = 48;
constexpr int C_  = 64;
constexpr int NH_ = 4;
constexpr int KD_ = 64;
constexpr int S_  = H_ * W_;     // 2304
constexpr int NK_ = NH_ * KD_;   // 256

DEV unsigned pkh(float a, float b) {
    union { _Float16 h[2]; unsigned u; } x;
    x.h[0] = (_Float16)a;
    x.h[1] = (_Float16)b;
    return x.u;
}

// xor-16 half-wave swap via v_permlanex16_b32 (VALU lane permute, no LDS
// pipe / DScnt wait like ds_bpermute). Identity nibble selects swap rows.
DEV unsigned xor16_u32(unsigned x) {
    return (unsigned)__builtin_amdgcn_permlanex16(
        (int)x, (int)x, 0x76543210, 0xfedcba98, false, false);
}
DEV float xor16_f32(float x) {
    union { float f; int i; } a, r;
    a.f = x;
    r.i = __builtin_amdgcn_permlanex16(a.i, a.i, 0x76543210, 0xfedcba98,
                                       false, false);
    return r.f;
}

// -------------------------------------------------------------------------
// Stage 1: q_in/kv_in = (out|inp) + pos;  QKV projections -> f16 workspace.
// Q, K stored [b*NH+n][s][kd] row-major; V stored transposed [b*NH+n][kd][s].
// -------------------------------------------------------------------------
DEV void dotcol(const float (*in)[C_], const float* __restrict__ Wmat,
                int col, float* acc) {
#pragma unroll
    for (int r = 0; r < 16; ++r) acc[r] = 0.f;
    for (int c = 0; c < C_; ++c) {
        float w = Wmat[c * NK_ + col];
#pragma unroll
        for (int r = 0; r < 16; ++r) acc[r] = fmaf(in[r][c], w, acc[r]);
    }
}

__global__ void __launch_bounds__(256) qkv_proj_kernel(
    const float* __restrict__ inp,  const float* __restrict__ outp,
    const float* __restrict__ posy, const float* __restrict__ posx,
    const float* __restrict__ Wq,   const float* __restrict__ bq,
    const float* __restrict__ Wk,   const float* __restrict__ bk,
    const float* __restrict__ Wv,   const float* __restrict__ bv,
    _Float16* __restrict__ Qb, _Float16* __restrict__ Kb,
    _Float16* __restrict__ Vtb)
{
    __shared__ float qin[16][C_];
    __shared__ float kvin[16][C_];
    const int t    = threadIdx.x;
    const int tile = blockIdx.x;             // B_*S_/16 tiles
    const int b    = tile / (S_ / 16);
    const int s0   = (tile % (S_ / 16)) * 16;
    const int base = (b * S_ + s0) * C_;
#pragma unroll
    for (int k = 0; k < 4; ++k) {
        int idx = t + k * 256;
        int r = idx >> 6, c = idx & 63;
        int s = s0 + r;
        int y = s / W_, x = s - y * W_;
        float p = posy[y * C_ + c] + posx[x * C_ + c];
        qin[r][c]  = outp[base + idx] + p;
        kvin[r][c] = inp[base + idx] + p;
    }
    __syncthreads();

    const int n  = t >> 6;
    const int kd = t & 63;
    const int bn = b * NH_ + n;
    float a[16];

    // Q (pre-scaled by 1/sqrt(KD) = 0.125)
    dotcol(qin, Wq, t, a);
    {
        float bb = bq[t];
#pragma unroll
        for (int r = 0; r < 16; ++r)
            Qb[(bn * S_ + s0 + r) * KD_ + kd] = (_Float16)((a[r] + bb) * 0.125f);
    }
    // K
    dotcol(kvin, Wk, t, a);
    {
        float bb = bk[t];
#pragma unroll
        for (int r = 0; r < 16; ++r)
            Kb[(bn * S_ + s0 + r) * KD_ + kd] = (_Float16)(a[r] + bb);
    }
    // V (transposed: [bn][kd][s])
    dotcol(kvin, Wv, t, a);
    {
        float bb = bv[t];
#pragma unroll
        for (int r = 0; r < 16; ++r)
            Vtb[(bn * KD_ + kd) * S_ + s0 + r] = (_Float16)(a[r] + bb);
    }
}

// -------------------------------------------------------------------------
// Stage 2: flash attention, one wave per (b, head, 16-query tile).
// Computes S^T = K * Q^T and ctx^T = V^T * P^T so that every WMMA operand
// fragment is a contiguous 16B load; P^T C->B relayout needs 8 permlanex16.
// -------------------------------------------------------------------------
__global__ void __launch_bounds__(256) flash_attn_kernel(
    const _Float16* __restrict__ Qb, const _Float16* __restrict__ Kb,
    const _Float16* __restrict__ Vtb, float* __restrict__ ctx)
{
    const int wave = blockIdx.x * (blockDim.x >> 5) + (threadIdx.x >> 5);
    const int lane = threadIdx.x & 31;
    const int qt   = wave % (S_ / 16);
    const int bn   = wave / (S_ / 16);

    const _Float16* Qp = Qb + ((size_t)bn * S_ + qt * 16) * KD_;
    const _Float16* Kp = Kb + (size_t)bn * S_ * KD_;
    const _Float16* Vp = Vtb + (size_t)bn * KD_ * S_;

    const int  qsel  = lane & 15;
    const bool lo    = lane < 16;
    const int  khalf = lo ? 0 : 16;   // V^T A-fragment K offset (lanes 16-31: K=16..31)
    const int  base8 = lo ? 0 : 8;    // A-fragment K offset (lanes 16-31: K=8..15,24..31)

    // Q^T B-fragments for kd chunks {0..31},{32..63}: contiguous row-major reads.
    V16H qf[2];
#pragma unroll
    for (int c = 0; c < 2; ++c) {
        const _Float16* p = Qp + qsel * KD_ + c * 32 + base8;
        qf[c].h[0] = *(const v8h*)(p);
        qf[c].h[1] = *(const v8h*)(p + 16);
    }

    v8f acc[4];
#pragma unroll
    for (int tt = 0; tt < 4; ++tt)
#pragma unroll
        for (int i = 0; i < 8; ++i) acc[tt][i] = 0.f;
    float m_run = -1e30f, l_run = 0.f;

    for (int it = 0; it < S_ / 32; ++it) {
        const int kb = it * 32;

        // K A-fragments: 2 key subtiles x 2 kd chunks.
        V16H kf[2][2];
#pragma unroll
        for (int sub = 0; sub < 2; ++sub)
#pragma unroll
            for (int c = 0; c < 2; ++c) {
                const _Float16* p =
                    Kp + (kb + sub * 16 + qsel) * KD_ + c * 32 + base8;
                kf[sub][c].h[0] = *(const v8h*)(p);
                kf[sub][c].h[1] = *(const v8h*)(p + 16);
            }

        // S^T = K * Q^T  (rows = keys, cols = queries), accumulate over kd.
        v8f st[2];
#pragma unroll
        for (int sub = 0; sub < 2; ++sub) {
            v8f z = {0.f, 0.f, 0.f, 0.f, 0.f, 0.f, 0.f, 0.f};
            v8f s1 = __builtin_amdgcn_wmma_f32_16x16x32_f16(
                false, kf[sub][0].v, false, qf[0].v, (short)0, z, false, false);
            st[sub] = __builtin_amdgcn_wmma_f32_16x16x32_f16(
                false, kf[sub][1].v, false, qf[1].v, (short)0, s1, false, false);
        }

        // Online softmax over the key axis (per query column; lane-half pair
        // l and l+16 share a query and split the 32 keys).
        float mloc = st[0][0];
#pragma unroll
        for (int i = 0; i < 8; ++i) {
            mloc = fmaxf(mloc, st[0][i]);
            mloc = fmaxf(mloc, st[1][i]);
        }
        float mtile = fmaxf(mloc, xor16_f32(mloc));
        float mnew  = fmaxf(m_run, mtile);
        float corr  = __expf(m_run - mnew);

        float e0[8], e1[8], sl = 0.f;
#pragma unroll
        for (int i = 0; i < 8; ++i) {
            e0[i] = __expf(st[0][i] - mnew);
            e1[i] = __expf(st[1][i] - mnew);
            sl += e0[i] + e1[i];
        }
        l_run = l_run * corr + (sl + xor16_f32(sl));
        m_run = mnew;
#pragma unroll
        for (int tt = 0; tt < 4; ++tt)
#pragma unroll
            for (int i = 0; i < 8; ++i) acc[tt][i] *= corr;

        // Re-layout P^T from C-fragment (2x 16keys x 16q) into a B-fragment
        // (32keys x 16q): pack to f16 pairs, swap lane halves for keys 8..15
        // and 16..23 with v_permlanex16.
        unsigned h0[4], h1[4];
#pragma unroll
        for (int j = 0; j < 4; ++j) {
            h0[j] = pkh(e0[2 * j], e0[2 * j + 1]);
            h1[j] = pkh(e1[2 * j], e1[2 * j + 1]);
        }
        V16H pf;
#pragma unroll
        for (int j = 0; j < 4; ++j) {
            unsigned x0 = xor16_u32(h0[j]);
            unsigned x1 = xor16_u32(h1[j]);
            pf.u[j]     = lo ? h0[j] : x1;
            pf.u[j + 4] = lo ? x0 : h1[j];
        }

        // ctx^T += V^T * P^T : 4 kd tiles of 16; V^T fragments come straight
        // from the transposed V buffer (contiguous along keys).
#pragma unroll
        for (int tt = 0; tt < 4; ++tt) {
            const _Float16* p = Vp + (tt * 16 + qsel) * S_ + kb + khalf;
            V16H vf;
            vf.h[0] = *(const v8h*)(p);
            vf.h[1] = *(const v8h*)(p + 8);
            acc[tt] = __builtin_amdgcn_wmma_f32_16x16x32_f16(
                false, vf.v, false, pf.v, (short)0, acc[tt], false, false);
        }
    }

    // Normalize and write ctx as f32 [b][s][n][kd].
    const float inv  = __builtin_amdgcn_rcpf(l_run);
    const int   srow = qt * 16 + qsel;
    const int   b    = bn >> 2, n = bn & 3;
    float* op = ctx + ((size_t)(b * S_ + srow) * NH_ + n) * KD_;
#pragma unroll
    for (int tt = 0; tt < 4; ++tt) {
        float4 va = make_float4(acc[tt][0] * inv, acc[tt][1] * inv,
                                acc[tt][2] * inv, acc[tt][3] * inv);
        float4 vb = make_float4(acc[tt][4] * inv, acc[tt][5] * inv,
                                acc[tt][6] * inv, acc[tt][7] * inv);
        *(float4*)(op + tt * 16 + base8)     = va;
        *(float4*)(op + tt * 16 + base8 + 4) = vb;
    }
}

// -------------------------------------------------------------------------
// Stage 3: output projection o = ctx @ Wo + bo (fp32).
// -------------------------------------------------------------------------
__global__ void __launch_bounds__(256) out_proj_kernel(
    const float* __restrict__ ctx, const float* __restrict__ Wo,
    const float* __restrict__ bo, float* __restrict__ o)
{
    __shared__ float tileS[16][NK_];   // 16 KB
    const int t  = threadIdx.x;
    const int bk = blockIdx.x;
    const int b  = bk / (S_ / 16);
    const int s0 = (bk % (S_ / 16)) * 16;
    const float* cp = ctx + (size_t)(b * S_ + s0) * NK_;
#pragma unroll
    for (int k = 0; k < 16; ++k) {
        int idx = t + k * 256;
        tileS[idx >> 8][idx & 255] = cp[idx];
    }
    __syncthreads();

    const int c  = t & 63;
    const int r0 = t >> 6;   // 0..3
    float accv[4];
#pragma unroll
    for (int i = 0; i < 4; ++i) accv[i] = bo[c];
    for (int nk = 0; nk < NK_; ++nk) {
        float w = Wo[nk * C_ + c];
#pragma unroll
        for (int i = 0; i < 4; ++i)
            accv[i] = fmaf(tileS[r0 + i * 4][nk], w, accv[i]);
    }
#pragma unroll
    for (int i = 0; i < 4; ++i)
        o[(b * S_ + s0 + r0 + i * 4) * C_ + c] = accv[i];
}

// -------------------------------------------------------------------------
extern "C" void kernel_launch(void* const* d_in, const int* in_sizes, int n_in,
                              void* d_out, int out_size, void* d_ws,
                              size_t ws_size, hipStream_t stream)
{
    (void)in_sizes; (void)n_in; (void)out_size; (void)ws_size;
    const float* inp  = (const float*)d_in[0];
    const float* outp = (const float*)d_in[1];
    const float* posy = (const float*)d_in[2];
    const float* posx = (const float*)d_in[3];
    const float* Wq   = (const float*)d_in[4];
    const float* bq   = (const float*)d_in[5];
    const float* Wk   = (const float*)d_in[6];
    const float* bk   = (const float*)d_in[7];
    const float* Wv   = (const float*)d_in[8];
    const float* bv   = (const float*)d_in[9];
    const float* Wo   = (const float*)d_in[10];
    const float* bo   = (const float*)d_in[11];
    float* o = (float*)d_out;

    char* ws = (char*)d_ws;
    const size_t szQ = (size_t)B_ * NH_ * S_ * KD_ * sizeof(_Float16); // 4.7 MB
    _Float16* Qb  = (_Float16*)(ws);
    _Float16* Kb  = (_Float16*)(ws + szQ);
    _Float16* Vtb = (_Float16*)(ws + 2 * szQ);
    float*    ctx = (float*)(ws + 3 * szQ);

    qkv_proj_kernel<<<dim3(B_ * S_ / 16), dim3(256), 0, stream>>>(
        inp, outp, posy, posx, Wq, bq, Wk, bk, Wv, bv, Qb, Kb, Vtb);
    flash_attn_kernel<<<dim3(B_ * NH_ * (S_ / 16) / 8), dim3(256), 0, stream>>>(
        Qb, Kb, Vtb, ctx);
    out_proj_kernel<<<dim3(B_ * S_ / 16), dim3(256), 0, stream>>>(
        ctx, Wo, bo, o);
}
